// InfoNCELoss_gradcache_90701119357443
// MI455X (gfx1250) — matrix operand order repaired and checked by reading, not compiled, not run on hardware
//
#include <hip/hip_runtime.h>
#include <hip/hip_bf16.h>

typedef __attribute__((ext_vector_type(16))) __bf16 v16bf;
typedef __attribute__((ext_vector_type(8)))  __bf16 v8bf;
typedef __attribute__((ext_vector_type(4)))  __bf16 v4bf;
typedef __attribute__((ext_vector_type(8)))  float  v8f;

constexpr int   DIM    = 128;
constexpr float INV_T  = 14.285714285714286f;    // 1 / 0.07
constexpr float EXP2_SCALE = 20.60992915555662f; // log2(e) / 0.07
constexpr int   JSPLIT = 8;                      // j-dimension grid split

// Raw v_exp_f32 (args here are >= -21, so no subnormal-range fixup needed).
__device__ __forceinline__ float fast_exp2(float x) {
#if __has_builtin(__builtin_amdgcn_exp2f)
  return __builtin_amdgcn_exp2f(x);
#else
  return exp2f(x);
#endif
}

// ---------------------------------------------------------------------------
// Kernel 1: row L2-normalize q and d (fp32 math), emit bf16 copies for WMMA
// and exact fp32 diagonal dot <q_i, d_i> for the numerator.
// One wave (32 lanes) per row; lane handles 4 contiguous floats (float4).
// ---------------------------------------------------------------------------
__global__ __launch_bounds__(256) void normalize_kernel(
    const float* __restrict__ q, const float* __restrict__ d,
    __bf16* __restrict__ qn, __bf16* __restrict__ dn,
    float* __restrict__ rowdiag, int n) {
  const int wave = threadIdx.x >> 5;
  const int lane = threadIdx.x & 31;
  const int row  = blockIdx.x * 8 + wave;
  if (row >= n) return;

  const float4 qv = ((const float4*)(q + (size_t)row * DIM))[lane];
  const float4 dv = ((const float4*)(d + (size_t)row * DIM))[lane];

  float qq = qv.x*qv.x + qv.y*qv.y + qv.z*qv.z + qv.w*qv.w;
  float dd = dv.x*dv.x + dv.y*dv.y + dv.z*dv.z + dv.w*dv.w;
  float qd = qv.x*dv.x + qv.y*dv.y + qv.z*dv.z + qv.w*dv.w;
  #pragma unroll
  for (int m = 1; m < 32; m <<= 1) {
    qq += __shfl_xor(qq, m, 32);
    dd += __shfl_xor(dd, m, 32);
    qd += __shfl_xor(qd, m, 32);
  }
  const float rq = rsqrtf(qq);
  const float rd = rsqrtf(dd);

  v4bf qb, db;
  qb[0] = (__bf16)(qv.x * rq); qb[1] = (__bf16)(qv.y * rq);
  qb[2] = (__bf16)(qv.z * rq); qb[3] = (__bf16)(qv.w * rq);
  db[0] = (__bf16)(dv.x * rd); db[1] = (__bf16)(dv.y * rd);
  db[2] = (__bf16)(dv.z * rd); db[3] = (__bf16)(dv.w * rd);
  ((v4bf*)(qn + (size_t)row * DIM))[lane] = qb;
  ((v4bf*)(dn + (size_t)row * DIM))[lane] = db;

  if (lane == 0) rowdiag[row] = qd * rq * rd;
}

// ---------------------------------------------------------------------------
// One GEMM strip: this wave's two A tiles vs jt_cnt 16-column B tiles,
// exp(s/t) accumulated into per-lane row partial sums.  Ping-pong B buffers
// (ba/bb), j-loop unrolled x2: no register copies, prefetch loads overlap the
// other buffer's 8 WMMAs + exp tail.  Requires jt_cnt even.
// ---------------------------------------------------------------------------
__device__ __forceinline__ void gemm_exp_strip(
    const __bf16* __restrict__ B, size_t jbase, size_t lane_off, int jt_cnt,
    const v16bf a0[4], const v16bf a1[4], float racc0[8], float racc1[8]) {
  const __bf16* bcol = B + jbase + lane_off;

  v16bf ba[4], bb[4];
  #pragma unroll
  for (int ks = 0; ks < 4; ++ks) ba[ks] = *(const v16bf*)(bcol + ks * 32);

  for (int it = 0; it < jt_cnt; it += 2) {       // uniform scalar trip count
    // prefetch tile it+1 into bb (always valid: jt_cnt even)
    #pragma unroll
    for (int ks = 0; ks < 4; ++ks)
      bb[ks] = *(const v16bf*)(bcol + 16 * DIM + ks * 32);

    v8f c0 = {}, c1 = {};
    #pragma unroll
    for (int ks = 0; ks < 4; ++ks) {
      c0 = __builtin_amdgcn_wmma_f32_16x16x32_bf16(false, a0[ks], false, ba[ks],
                                                   (short)0, c0, false, false);
      c1 = __builtin_amdgcn_wmma_f32_16x16x32_bf16(false, a1[ks], false, ba[ks],
                                                   (short)0, c1, false, false);
    }
    #pragma unroll
    for (int r = 0; r < 8; ++r) {
      racc0[r] += fast_exp2(c0[r] * EXP2_SCALE); // exp(s/t) = 2^(s*log2e/t)
      racc1[r] += fast_exp2(c1[r] * EXP2_SCALE);
    }

    // prefetch tile it+2 into ba (clamped on the final pair; uniform select)
    const __bf16* bn2 = bcol + ((it + 2 < jt_cnt) ? 2 * 16 * DIM : 0);
    #pragma unroll
    for (int ks = 0; ks < 4; ++ks) ba[ks] = *(const v16bf*)(bn2 + ks * 32);

    v8f e0 = {}, e1 = {};
    #pragma unroll
    for (int ks = 0; ks < 4; ++ks) {
      e0 = __builtin_amdgcn_wmma_f32_16x16x32_bf16(false, a0[ks], false, bb[ks],
                                                   (short)0, e0, false, false);
      e1 = __builtin_amdgcn_wmma_f32_16x16x32_bf16(false, a1[ks], false, bb[ks],
                                                   (short)0, e1, false, false);
    }
    #pragma unroll
    for (int r = 0; r < 8; ++r) {
      racc0[r] += fast_exp2(e0[r] * EXP2_SCALE);
      racc1[r] += fast_exp2(e1[r] * EXP2_SCALE);
    }

    bcol += 2 * 16 * DIM;
  }
}

// ---------------------------------------------------------------------------
// Kernel 2: fused  rowsum_i = sum_j exp(<qn_i,dn_j>/t) + sum_j exp(<qn_i,qn_j>/t)
// via v_wmma_f32_16x16x32_bf16.  Each wave owns TWO register-resident 16-row
// A tiles; block = 8 waves * 32 rows = 256 rows; grid = (N/256, JSPLIT); all
// waves of a block share the same B range -> WGP-cache hits.
// ---------------------------------------------------------------------------
__global__ __launch_bounds__(256) void scores_kernel(
    const __bf16* __restrict__ qn, const __bf16* __restrict__ dn,
    float* __restrict__ rowsum, int n) {
  const int lane = threadIdx.x & 31;
  const int wave = threadIdx.x >> 5;
  const int hi   = lane >> 4;            // K-half select (0 or 1)
  const int mrow = lane & 15;
  const int base_row = blockIdx.x * 256 + wave * 32;   // two 16-row tiles

  // ---- A: 2 tiles x 16 rows x 128 K of normalized q, WMMA bf16 layout ----
  v16bf a0[4], a1[4];
  {
    const __bf16* ar0 = qn + (size_t)(base_row + mrow) * DIM;
    const __bf16* ar1 = ar0 + 16 * DIM;
    #pragma unroll
    for (int ks = 0; ks < 4; ++ks) {
      v8bf x0 = *(const v8bf*)(ar0 + ks * 32 + hi * 16);
      v8bf x1 = *(const v8bf*)(ar0 + ks * 32 + hi * 16 + 8);
      a0[ks] = __builtin_shufflevector(x0, x1, 0,1,2,3,4,5,6,7,8,9,10,11,12,13,14,15);
      v8bf y0 = *(const v8bf*)(ar1 + ks * 32 + hi * 16);
      v8bf y1 = *(const v8bf*)(ar1 + ks * 32 + hi * 16 + 8);
      a1[ks] = __builtin_shufflevector(y0, y1, 0,1,2,3,4,5,6,7,8,9,10,11,12,13,14,15);
    }
  }

  float racc0[8], racc1[8];
  #pragma unroll
  for (int r = 0; r < 8; ++r) { racc0[r] = 0.f; racc1[r] = 0.f; }

  const int jt_cnt = (n / 16) / JSPLIT;          // uniform scalar trip count
  const size_t jbase = (size_t)blockIdx.y * jt_cnt * 16 * DIM;
  const size_t lane_off = (size_t)mrow * DIM + hi * 16;

  gemm_exp_strip(dn, jbase, lane_off, jt_cnt, a0, a1, racc0, racc1); // q.d^T
  gemm_exp_strip(qn, jbase, lane_off, jt_cnt, a0, a1, racc0, racc1); // q.q^T

  // Reduce the 16 columns held across lanes of each half-wave.
  // C/D layout: vgpr r -> row (base + r) on lanes 0-15, (base + 8 + r) on 16-31.
  #pragma unroll
  for (int r = 0; r < 8; ++r) {
    float v0 = racc0[r], v1 = racc1[r];
    #pragma unroll
    for (int m = 1; m < 16; m <<= 1) {
      v0 += __shfl_xor(v0, m, 32);
      v1 += __shfl_xor(v1, m, 32);
    }
    if ((lane & 15) == 0) {
      atomicAdd(&rowsum[base_row + r + hi * 8], v0);
      atomicAdd(&rowsum[base_row + 16 + r + hi * 8], v1);
    }
  }
}

// ---------------------------------------------------------------------------
// Kernel 3: per-row loss terms, wave-reduced, atomically accumulated.
// loss_i = diag_i/t - log(rowsum_i - exp(1/t))
// ---------------------------------------------------------------------------
__global__ __launch_bounds__(256) void finalize_partial(
    const float* __restrict__ rowdiag, const float* __restrict__ rowsum,
    float* __restrict__ acc, int n) {
  const int i = blockIdx.x * blockDim.x + threadIdx.x;
  float v = 0.f;
  if (i < n) {
    const float denom = rowsum[i] - exp2f(EXP2_SCALE);  // remove q_i.q_i term
    v = rowdiag[i] * INV_T - __logf(denom);
  }
  #pragma unroll
  for (int m = 1; m < 32; m <<= 1) v += __shfl_xor(v, m, 32);
  if ((threadIdx.x & 31) == 0) atomicAdd(acc, v);
}

__global__ void finalize_write(const float* __restrict__ acc,
                               float* __restrict__ out, int n) {
  out[0] = -acc[0] / (float)n;
}

// ---------------------------------------------------------------------------
extern "C" void kernel_launch(void* const* d_in, const int* in_sizes, int n_in,
                              void* d_out, int out_size, void* d_ws, size_t ws_size,
                              hipStream_t stream) {
  const float* q = (const float*)d_in[0];
  const float* d = (const float*)d_in[1];
  const int n = in_sizes[0] / DIM;               // 16384

  char* ws = (char*)d_ws;
  __bf16* qn      = (__bf16*)ws;                                  // n*DIM*2 B
  __bf16* dn      = (__bf16*)(ws + (size_t)n * DIM * 2);          // n*DIM*2 B
  float*  rowdiag = (float*)(ws + (size_t)n * DIM * 4);           // n*4 B
  float*  rowsum  = rowdiag + n;                                  // n*4 B
  float*  acc     = rowsum + n;                                   // 4 B

  // zero rowsum + scalar accumulator (graph-capture-safe async memset)
  hipMemsetAsync(rowsum, 0, (size_t)(n + 1) * sizeof(float), stream);

  normalize_kernel<<<n / 8, 256, 0, stream>>>(q, d, qn, dn, rowdiag, n);

  dim3 grid2(n / 256, JSPLIT);
  scores_kernel<<<grid2, 256, 0, stream>>>(qn, dn, rowsum, n);

  finalize_partial<<<(n + 255) / 256, 256, 0, stream>>>(rowdiag, rowsum, acc, n);
  finalize_write<<<1, 1, 0, stream>>>(acc, (float*)d_out, n);
}
